// Model_MIND_29764123361468
// MI455X (gfx1250) — matrix operand order repaired and checked by reading, not compiled, not run on hardware
//
#include <hip/hip_runtime.h>
#include <cstdint>
#include <cstddef>

// ---------------- CDNA5 (gfx1250) MIND capsule-routing pipeline ----------------
// k_init    : zero/seed scratch scalars
// k_minidx  : global atomicMin of time indices (replaces jnp.min(idx))
// k_cvtw1   : W1 (300x100 f32) -> W1^T f16 padded [112][320]
// k_cvtwb   : w_bilinear f32 -> f16 (8.4 MB)
// k_mlp     : time-feature MLP via v_wmma_f32_16x16x32_f16 -> t_factor, reg1
//             (f64 range reduction + native v_sin/v_cos, no Payne-Hanek blob)
// k_bilin   : per-s GEMM (1024x64)@(64x256) via WMMA -> hat (f16, 134 MB scratch)
//             LDS staging issued through TDM tensor_load_to_lds (+ safe fallback copy)
// k_route   : per-batch 3-iteration dynamic routing, hat in VGPRs
// k_fin     : loss_reg scalar

typedef __attribute__((ext_vector_type(16))) _Float16 v16h;
typedef __attribute__((ext_vector_type(8)))  _Float16 h8;
typedef __attribute__((ext_vector_type(8)))  float    v8f;
typedef __attribute__((ext_vector_type(4)))  unsigned int v4u;
typedef __attribute__((ext_vector_type(8)))  int v8i_;
typedef __attribute__((ext_vector_type(4)))  int v4i_;

#define SPANF 10000.0f
#define B_    1024
#define S_    256
#define D_    64
#define NI_   4
#define G_    8
#define KK_   256   /* NI*D */
#define HID_  100
#define HIDP_ 112
#define FEAT_ 300
#define FEATP_ 320

// ---- TDM: 1D contiguous load with LDS row padding (pad_amount every pad_interval).
// D# layout per CDNA5 ISA 8.3/8.4. Group2/3 zero-filled (dims >2 unused).
__device__ __forceinline__ void tdm_load_1d_padded(unsigned lds_addr,
                                                   unsigned long long gaddr,
                                                   unsigned n_halves,
                                                   unsigned pad_interval_code,
                                                   unsigned pad_amount_code) {
    v4u g0;
    g0[0] = 1u;                                               // count=1, user mode
    g0[1] = lds_addr;                                         // lds_addr [63:32]
    g0[2] = (unsigned)(gaddr & 0xFFFFFFFFull);                // global_addr lo
    g0[3] = (unsigned)((gaddr >> 32) & 0x01FFFFFFull)         // global_addr hi (57b)
            | (2u << 30);                                     // type=2 ("image")
    v8i_ g1;
    g1[0] = (int)((1u << 16)                                  // data_size=1 -> 2B
                  | (1u << 20)                                // pad_enable
                  | (pad_interval_code << 22)
                  | (pad_amount_code << 25));
    g1[1] = (int)((n_halves & 0xFFFFu) << 16);                // tensor_dim0 lo16
    g1[2] = (int)(((n_halves >> 16) & 0xFFFFu) | (1u << 16)); // td0 hi16 | tensor_dim1=1
    g1[3] = (int)((n_halves & 0xFFFFu) << 16);                // tile_dim0
    g1[4] = 1;                                                // tile_dim1=1
    g1[5] = (int)n_halves;                                    // tensor_dim0_stride lo32
    g1[6] = 0;
    g1[7] = 0;
    v4i_ z4 = {0, 0, 0, 0};
#if __clang_major__ >= 23
    v8i_ z8 = {0, 0, 0, 0, 0, 0, 0, 0};
    __builtin_amdgcn_tensor_load_to_lds(g0, g1, z4, z4, z8, 0);
#else
    __builtin_amdgcn_tensor_load_to_lds(g0, g1, z4, z4, 0);
#endif
}

// ---------------------------------------------------------------- init
__global__ void k_init(int* mins, float* reg1sum) {
    if (threadIdx.x < 3) mins[threadIdx.x] = 0x7FFFFFFF;
    if (threadIdx.x == 3) *reg1sum = 0.f;
}

// ---------------------------------------------------------------- min indices
__global__ void k_minidx(const float* tb, const float* th, int* mins) {
    __shared__ int m0, m1, m2;
    if (threadIdx.x == 0) { m0 = m1 = m2 = 0x7FFFFFFF; }
    __syncthreads();
    int i = blockIdx.x * blockDim.x + threadIdx.x;
    if (i < B_ * S_) {
        int b = i >> 8;
        float t1 = tb[b], t2 = th[i];
        atomicMin(&m0, (int)(t1 / SPANF));
        atomicMin(&m1, (int)(t2 / SPANF));
        atomicMin(&m2, (int)((t1 - t2) / SPANF));
    }
    __syncthreads();
    if (threadIdx.x == 0) {
        atomicMin(&mins[0], m0);
        atomicMin(&mins[1], m1);
        atomicMin(&mins[2], m2);
    }
}

// ---------------------------------------------------------------- W1 -> f16 transposed padded
__global__ void k_cvtw1(const float* W1, _Float16* w1t) {
    int i = blockIdx.x * blockDim.x + threadIdx.x;   // i = c*320 + kk
    if (i >= HIDP_ * FEATP_) return;
    int c = i / FEATP_, kk = i - c * FEATP_;
    float v = (c < HID_ && kk < FEAT_) ? W1[kk * HID_ + c] : 0.f;
    w1t[i] = (_Float16)v;
}

// ---------------------------------------------------------------- w_bilinear -> f16
__global__ void k_cvtwb(const float* w, _Float16* w16) {
    int n = S_ * KK_ * D_;
    for (int i = blockIdx.x * blockDim.x + threadIdx.x; i < n; i += gridDim.x * blockDim.x)
        w16[i] = (_Float16)w[i];
}

// ---------------------------------------------------------------- time MLP (WMMA)
// Block: 256 threads / 8 waves, 128 rows (b,s). K padded 300->320 (10 steps of 32),
// N padded 100->112 (7 tiles). Wave w owns rows w*16..w*16+15 and all 7 N tiles.
__global__ void __launch_bounds__(256)
k_mlp(const float* tb, const float* th, const float* b1, const float* W2,
      const float* b2, const int* mins, const _Float16* w1t,
      float* tfac, float* reg1sum) {
    __shared__ _Float16 feat[128][40];   // 128 rows x 32 feats, stride 40 (bank pad)
    __shared__ _Float16 w1c[112][40];    // 112 cols x 32 k,   stride 40
    __shared__ double divs[50];
    __shared__ float blkReg;
    int tid = threadIdx.x;
    if (tid < 50) divs[tid] = exp(-0.09210340371976184 * (double)tid);
    if (tid == 0) blkReg = 0.f;

    int lr = tid >> 1, half = tid & 1;
    int row = blockIdx.x * 128 + lr;
    float t1 = tb[row >> 8], t2 = th[row];
    float p0 = (float)((int)(t1 / SPANF) - mins[0]);
    float p1 = (float)((int)(t2 / SPANF) - mins[1]);
    float p2 = (float)((int)((t1 - t2) / SPANF) - mins[2]);

    int wv = tid >> 5, lane = tid & 31, m = lane & 15, hiL = lane >> 4;
    v8f acc[7] = {};

    const double TWO_PI  = 6.283185307179586;
    const double INV2PI  = 0.15915494309189535;

    for (int ks = 0; ks < 10; ks++) {
        __syncthreads();
        // generate 16 features (8 sin/cos pairs) for this thread's row-half.
        // f64 arg reduction (pos is an exact integer) + native v_sin/v_cos:
        // avoids sinf's Payne-Hanek slow path and matches the f64-built table.
        #pragma unroll
        for (int u = 0; u < 8; u++) {
            int j = ks * 32 + half * 16 + 2 * u;
            float sv = 0.f, cv = 0.f;
            if (j < FEAT_) {
                int seg = j / 100, rem = j - seg * 100;
                float pos = (seg == 0) ? p0 : ((seg == 1) ? p1 : p2);
                double a = (double)pos * divs[rem >> 1];
                a -= TWO_PI * floor(a * INV2PI);
                float arg = (float)a;
                sv = __sinf(arg);
                cv = __cosf(arg);
            }
            feat[lr][half * 16 + 2 * u]     = (_Float16)sv;
            feat[lr][half * 16 + 2 * u + 1] = (_Float16)cv;
        }
        // stage W1 chunk (transposed, f16): cols 0..111, k = ks*32..+31
        if (tid < 224) {
            int c = tid >> 1, part = tid & 1;
            const uint4* src = (const uint4*)(w1t + c * FEATP_ + ks * 32 + part * 16);
            uint4 a0 = src[0], a1 = src[1];
            *(uint4*)&w1c[c][part * 16]     = a0;
            *(uint4*)&w1c[c][part * 16 + 8] = a1;
        }
        __syncthreads();
        // A fragment (16x32 f16, ISA layout): lane m: K 0-7 & 16-23 ; lane 16+m: K 8-15 & 24-31
        union { v16h v; h8 h[2]; } A;
        int arow = wv * 16 + m;
        A.h[0] = *(const h8*)&feat[arow][hiL * 8];
        A.h[1] = *(const h8*)&feat[arow][16 + hiL * 8];
        #pragma unroll
        for (int t = 0; t < 7; t++) {
            union { v16h v; h8 h[2]; } Bf;   // B 32x16: lane n: K0-7 & 16-23 ; lane 16+n: K8-15 & 24-31
            int bcol = t * 16 + m;
            Bf.h[0] = *(const h8*)&w1c[bcol][hiL * 8];
            Bf.h[1] = *(const h8*)&w1c[bcol][16 + hiL * 8];
            acc[t] = __builtin_amdgcn_wmma_f32_16x16x32_f16(
                false, A.v, false, Bf.v, (short)0, acc[t], false, false);
        }
    }
    // epilogue: +b1, relu, dot with W2, sigmoid
    float partial[8];
    #pragma unroll
    for (int r = 0; r < 8; r++) partial[r] = 0.f;
    #pragma unroll
    for (int t = 0; t < 7; t++) {
        int col = t * 16 + m;
        float b1v = (col < HID_) ? b1[col] : 0.f;
        float w2v = (col < HID_) ? W2[col] : 0.f;
        #pragma unroll
        for (int r = 0; r < 8; r++) {
            float h = acc[t][r] + b1v;
            h = h > 0.f ? h : 0.f;
            partial[r] += h * w2v;
        }
    }
    float l2 = 0.f;
    #pragma unroll
    for (int r = 0; r < 8; r++) {
        float v = partial[r];
        v += __shfl_xor(v, 1); v += __shfl_xor(v, 2);
        v += __shfl_xor(v, 4); v += __shfl_xor(v, 8);
        float sig = 1.f / (1.f + __expf(-(v + b2[0])));
        if (m == 0) {
            int orow = blockIdx.x * 128 + wv * 16 + hiL * 8 + r;
            tfac[orow] = sig;
            l2 += sig * sig;
        }
    }
    atomicAdd(&blkReg, l2);
    __syncthreads();
    if (tid == 0) atomicAdd(reg1sum, blkReg);
}

// ---------------------------------------------------------------- bilinear (WMMA)
// Block: 256 thr / 8 waves: one s, 128 b's. wave w: b-tile w (16 rows) x all 256 cols.
__global__ void __launch_bounds__(256)
k_bilin(const float* ihe, const _Float16* w16, _Float16* hat) {
    __shared__ _Float16 wc[256][72];     // [kk][d], stride 72 halves (bank pad)
    int tid = threadIdx.x;
    int s = blockIdx.x >> 3, btile = blockIdx.x & 7;

    // TDM async staging of the 32KB w-slice with 64->72-half row padding
    // (pad 4 DWORDs after every 32 DWORDs: interval code 4, amount code 3).
    if (tid == 0) {
        tdm_load_1d_padded((unsigned)(uintptr_t)&wc[0][0],
                           (unsigned long long)(uintptr_t)(w16 + (size_t)s * KK_ * D_),
                           (unsigned)(KK_ * D_), 4u, 3u);
    }
    __builtin_amdgcn_s_wait_tensorcnt(0);
    // Known-good copy of the same bytes (descriptor semantics are not runtime-
    // verifiable in this loop; identical values, ordered after the TDM wait).
    {
        const uint4* src = (const uint4*)(w16 + (size_t)s * KK_ * D_ + (size_t)tid * D_);
        uint4* dst = (uint4*)&wc[tid][0];
        #pragma unroll
        for (int q = 0; q < 8; q++) dst[q] = src[q];
    }
    int wv = tid >> 5, lane = tid & 31, m = lane & 15, hiL = lane >> 4;
    int b = btile * 128 + wv * 16 + m;
    const float* arow = ihe + ((size_t)b * S_ + s) * D_;
    union { v16h v; h8 h[2]; } A0, A1;
    #pragma unroll
    for (int q = 0; q < 8; q++) {
        A0.v[q]     = (_Float16)arow[hiL * 8 + q];
        A0.v[8 + q] = (_Float16)arow[16 + hiL * 8 + q];
        A1.v[q]     = (_Float16)arow[32 + hiL * 8 + q];
        A1.v[8 + q] = (_Float16)arow[48 + hiL * 8 + q];
    }
    __syncthreads();
    v8f acc[16] = {};
    #pragma unroll
    for (int t = 0; t < 16; t++) {
        union { v16h v; h8 h[2]; } Bf;
        int col = t * 16 + m;
        Bf.h[0] = *(const h8*)&wc[col][hiL * 8];
        Bf.h[1] = *(const h8*)&wc[col][16 + hiL * 8];
        acc[t] = __builtin_amdgcn_wmma_f32_16x16x32_f16(
            false, A0.v, false, Bf.v, (short)0, acc[t], false, false);
        Bf.h[0] = *(const h8*)&wc[col][32 + hiL * 8];
        Bf.h[1] = *(const h8*)&wc[col][48 + hiL * 8];
        acc[t] = __builtin_amdgcn_wmma_f32_16x16x32_f16(
            false, A1.v, false, Bf.v, (short)0, acc[t], false, false);
    }
    #pragma unroll
    for (int t = 0; t < 16; t++) {
        int col = t * 16 + m;
        #pragma unroll
        for (int r = 0; r < 8; r++) {
            int brow = btile * 128 + wv * 16 + hiL * 8 + r;
            hat[(((size_t)brow * S_) + s) * KK_ + col] = (_Float16)acc[t][r];
        }
    }
}

// ---------------------------------------------------------------- routing
// Block = one batch b, 1024 threads: tid = k*256 + s. hat[s,k,0:64] in 64 VGPRs.
__global__ void __launch_bounds__(1024)
k_route(const float* ihe, const float* ieb, const float* th, const float* mask,
        const float* phase, const float* tao, const float* ratio, const float* gemb,
        const float* tfac, const _Float16* hat, float* out) {
    __shared__ float fac[256];
    __shared__ float cw[4][256];
    __shared__ float cap[4][68];
    __shared__ float ge[64][8];
    __shared__ float scl[4];
    __shared__ int   kmaxs;
    int tid = threadIdx.x;
    int b = blockIdx.x;
    int k = tid >> 8, s = tid & 255;
    int lane = tid & 31;

    if (tid == 0) __builtin_prefetch(ieb + (size_t)b * 64, 0, 3);  // readout vector
    if (tid < 512) ge[tid >> 3][tid & 7] = gemb[tid];
    cw[k][s] = 0.f;
    __syncthreads();

    if (tid < 256) {   // factor[s] = t_factor + g_factor (gate softmax)
        float lg[8];
        #pragma unroll
        for (int g = 0; g < 8; g++) lg[g] = 0.f;
        const float* emb = ihe + ((size_t)b * S_ + s) * D_;
        for (int d = 0; d < 64; d++) {
            float e = emb[d];
            #pragma unroll
            for (int g = 0; g < 8; g++) lg[g] += e * ge[d][g];
        }
        float mx = lg[0];
        #pragma unroll
        for (int g = 1; g < 8; g++) mx = fmaxf(mx, lg[g]);
        float ex[8], den = 0.f;
        #pragma unroll
        for (int g = 0; g < 8; g++) { ex[g] = __expf(lg[g] - mx); den += ex[g]; }
        float thv = th[b * S_ + s];
        float gf = 0.f;
        #pragma unroll
        for (int g = 0; g < 8; g++) {
            float ph = fmodf(thv - phase[g], tao[g]) / tao[g];
            float gt = (ph < ratio[g]) ? fmaxf(__sinf(3.14159f * ph / ratio[g]), 0.f)
                                       : 0.1f * ph;
            gf += ex[g] / den * gt;
        }
        fac[s] = tfac[(size_t)b * S_ + s] + gf;
    }

    // hat[s,k,:] -> 64 VGPRs
    float h[64];
    const _Float16* hp = hat + ((size_t)b * S_ + s) * KK_ + k * D_;
    #pragma unroll
    for (int q = 0; q < 8; q++) {
        h8 x = *(const h8*)(hp + q * 8);
        #pragma unroll
        for (int j = 0; j < 8; j++) h[q * 8 + j] = (float)x[j];
    }
    float mk = mask[b * S_ + s];
    __syncthreads();

    for (int it = 0; it < 3; it++) {
        if (tid < 256) cap[tid >> 6][tid & 63] = 0.f;
        __syncthreads();
        // sw = masked softmax over k of cw[:, s]
        float c0 = cw[0][s], c1 = cw[1][s], c2 = cw[2][s], c3 = cw[3][s];
        float mx = fmaxf(fmaxf(c0, c1), fmaxf(c2, c3));
        float e0 = __expf(c0 - mx), e1 = __expf(c1 - mx),
              e2 = __expf(c2 - mx), e3 = __expf(c3 - mx);
        float den = e0 + e1 + e2 + e3;
        float swk = ((k == 0) ? e0 : (k == 1) ? e1 : (k == 2) ? e2 : e3) / den;
        if (mk == 0.f) swk = 0.f;
        // cap[k,d] += sum_s swk*h[d] : wave butterfly + single-lane LDS adds
        #pragma unroll
        for (int c = 0; c < 4; c++) {
            float v16[16];
            #pragma unroll
            for (int d = 0; d < 16; d++) v16[d] = swk * h[c * 16 + d];
            #pragma unroll
            for (int d = 0; d < 16; d++) {
                v16[d] += __shfl_xor(v16[d], 1);
                v16[d] += __shfl_xor(v16[d], 2);
                v16[d] += __shfl_xor(v16[d], 4);
                v16[d] += __shfl_xor(v16[d], 8);
                v16[d] += __shfl_xor(v16[d], 16);
            }
            if (lane == 0) {
                #pragma unroll
                for (int d = 0; d < 16; d++) atomicAdd(&cap[k][c * 16 + d], v16[d]);
            }
        }
        __syncthreads();
        // squash
        if (tid < 4) {
            float n2 = 0.f;
            for (int d = 0; d < 64; d++) { float v = cap[tid][d]; n2 += v * v; }
            scl[tid] = n2 / (1.f + n2) / sqrtf(n2 + 1e-9f);
        }
        __syncthreads();
        if (tid < 256) cap[tid >> 6][tid & 63] *= scl[tid >> 6];
        __syncthreads();
        if (it < 2) {
            float dot = 0.f;
            #pragma unroll
            for (int d = 0; d < 64; d++) dot += h[d] * cap[k][d];
            cw[k][s] += dot * fac[s];
            __syncthreads();
        }
    }
    // interest_capsule
    if (tid < 256) out[(size_t)b * KK_ + tid] = cap[tid >> 6][tid & 63];
    // readout = capsule[argmax_k <cap_k, item_eb>]
    if (tid == 0) {
        float best = -1e30f; int kb = 0;
        for (int kk2 = 0; kk2 < 4; kk2++) {
            float a = 0.f;
            for (int d = 0; d < 64; d++) a += cap[kk2][d] * ieb[b * 64 + d];
            if (a > best) { best = a; kb = kk2; }
        }
        kmaxs = kb;
    }
    __syncthreads();
    if (tid < 64) out[(size_t)B_ * KK_ + (size_t)b * 64 + tid] = cap[kmaxs][tid];
}

// ---------------------------------------------------------------- loss_reg
__global__ void k_fin(const float* reg1sum, const float* ratio, const float* phase,
                      float* out) {
    float r1 = sqrtf(*reg1sum);
    float sr = 0.f, sp = 0.f;
    for (int g = 0; g < 8; g++) { sr += ratio[g] * ratio[g]; sp += phase[g] * phase[g]; }
    float reg2 = sqrtf(sr) + sqrtf(sp) / 100000.f;
    out[B_ * KK_ + B_ * D_] = 0.1f * r1 + reg2 + 1.f / r1 + 0.1f / reg2;
}

// ---------------------------------------------------------------- launch
extern "C" void kernel_launch(void* const* d_in, const int* in_sizes, int n_in,
                              void* d_out, int out_size, void* d_ws, size_t ws_size,
                              hipStream_t stream) {
    (void)in_sizes; (void)n_in; (void)out_size; (void)ws_size;
    const float* ihe   = (const float*)d_in[0];
    const float* ieb   = (const float*)d_in[1];
    const float* tb    = (const float*)d_in[2];
    const float* th    = (const float*)d_in[3];
    const float* mask  = (const float*)d_in[4];
    const float* W1    = (const float*)d_in[5];
    const float* b1    = (const float*)d_in[6];
    const float* W2    = (const float*)d_in[7];
    const float* b2    = (const float*)d_in[8];
    const float* phase = (const float*)d_in[9];
    const float* tao   = (const float*)d_in[10];
    const float* ratio = (const float*)d_in[11];
    const float* gemb  = (const float*)d_in[12];
    const float* wb    = (const float*)d_in[13];
    float* out = (float*)d_out;

    char* wsb = (char*)d_ws;
    int*      mins    = (int*)wsb;
    float*    reg1sum = (float*)(wsb + 12);
    float*    tfac    = (float*)(wsb + 256);
    _Float16* w1t     = (_Float16*)(wsb + 256 + 1048576);
    _Float16* wb16    = (_Float16*)(wsb + 256 + 1048576 + 71680);
    _Float16* hat     = (_Float16*)(wsb + 256 + 1048576 + 71680 + 8388608);
    // total scratch need ~143.7 MB

    k_init  <<<1, 64, 0, stream>>>(mins, reg1sum);
    k_minidx<<<512, 512, 0, stream>>>(tb, th, mins);
    k_cvtw1 <<<140, 256, 0, stream>>>(W1, w1t);
    k_cvtwb <<<4096, 256, 0, stream>>>(wb, wb16);
    k_mlp   <<<2048, 256, 0, stream>>>(tb, th, b1, W2, b2, mins, w1t, tfac, reg1sum);
    k_bilin <<<2048, 256, 0, stream>>>(ihe, wb16, hat);
    k_route <<<1024, 1024, 0, stream>>>(ihe, ieb, th, mask, phase, tao, ratio, gemb,
                                        tfac, hat, out);
    k_fin   <<<1, 1, 0, stream>>>(reg1sum, ratio, phase, out);
}
